// ParallelS4Layer_66846870995249
// MI455X (gfx1250) — compile-verified, matrix operand order checked by reading, not run on hardware
//
#include <hip/hip_runtime.h>

// ---------------------------------------------------------------------------
// ParallelS4Layer for MI455X (gfx1250, wave32)
//   x:[8,4096,512] f32 -> LN -> @Win+b -> S4 FFT conv (bidir, len 8192)
//   -> +D*u -> @Wout+b -> gelu(erf) -> +x
// GEMMs: bf16 WMMA (v_wmma_f32_16x16x32_bf16), 32x32 wave tile, f32 accum.
// Conv:  radix-2 DIF fwd / DIT inv FFT in 64KB LDS, bit-reversed-order
//        product; input gather via global_load_async_to_lds_b32 (ASYNCcnt).
// ---------------------------------------------------------------------------

#define D_MODEL 512
#define D_STATE 64
#define NCHUNK  128
#define BSZ     8
#define LSEQ    4096
#define NTOK    (BSZ * LSEQ)          // 32768 token rows
#define NFFT    8192                  // 2*L
#define HFFT    4096                  // NFFT/2 butterflies per stage

typedef __attribute__((ext_vector_type(16))) __bf16 v16bf;
typedef __attribute__((ext_vector_type(8)))  __bf16 v8bf;
typedef __attribute__((ext_vector_type(8)))  float  v8f;

union Frag16 { v16bf v; v8bf h[2]; };

// ---------------------------------------------------------------------------
// Weight convert: WT[n][k] = bf16(W[k][n])  (512x512)
// ---------------------------------------------------------------------------
__global__ __launch_bounds__(256) void k_weight_t(const float* __restrict__ W,
                                                  __bf16* __restrict__ WT) {
    int idx = blockIdx.x * 256 + threadIdx.x;       // over 512*512
    int k = idx >> 9, n = idx & 511;
    WT[(size_t)n * D_MODEL + k] = (__bf16)W[idx];
}

// ---------------------------------------------------------------------------
// LayerNorm -> bf16   (one block per token row, 256 thr, 2 elems/thr)
// ---------------------------------------------------------------------------
__global__ __launch_bounds__(256) void k_layernorm(const float* __restrict__ x,
                                                   const float* __restrict__ g,
                                                   const float* __restrict__ b,
                                                   __bf16* __restrict__ z) {
    __shared__ float red[256];
    __shared__ float s_mu, s_rs;
    const int tid = threadIdx.x;
    const size_t base = (size_t)blockIdx.x * D_MODEL;
    float v0 = x[base + tid];
    float v1 = x[base + tid + 256];
    red[tid] = v0 + v1;
    __syncthreads();
    for (int s = 128; s > 0; s >>= 1) {
        if (tid < s) red[tid] += red[tid + s];
        __syncthreads();
    }
    if (tid == 0) s_mu = red[0] * (1.0f / D_MODEL);
    __syncthreads();
    float mu = s_mu;
    float d0 = v0 - mu, d1 = v1 - mu;
    red[tid] = d0 * d0 + d1 * d1;
    __syncthreads();
    for (int s = 128; s > 0; s >>= 1) {
        if (tid < s) red[tid] += red[tid + s];
        __syncthreads();
    }
    if (tid == 0) s_rs = __frsqrt_rn(red[0] * (1.0f / D_MODEL) + 1e-5f);
    __syncthreads();
    float rs = s_rs;
    z[base + tid]       = (__bf16)(d0 * rs * g[tid]       + b[tid]);
    z[base + tid + 256] = (__bf16)(d1 * rs * g[tid + 256] + b[tid + 256]);
}

// ---------------------------------------------------------------------------
// WMMA GEMM core: C[M=NTOK, N=512] = A(bf16) x W^T(bf16)  (K=512)
// Block tile 128x64, 8 waves (4 M x 2 N), each wave a 32x32 tile:
// 2 A-frags x 2 B-frags -> 4 WMMAs per K-step (reuse both operands).
// Per-lane fragment layout (16-bit A 16x32):
//   element e -> K = (e/8)*16 + (lane/16)*8 + (e%8)  => two 16B loads per frag
// ---------------------------------------------------------------------------
__device__ __forceinline__ void gemm_core(const __bf16* __restrict__ A,
                                          const __bf16* __restrict__ WT,
                                          int row0, int col0, int lane,
                                          v8f acc[2][2]) {
    const int m = lane & 15;
    const int h = lane >> 4;
    const __bf16* a0 = A  + (size_t)(row0 + m)      * D_MODEL + h * 8;
    const __bf16* a1 = A  + (size_t)(row0 + 16 + m) * D_MODEL + h * 8;
    const __bf16* b0 = WT + (size_t)(col0 + m)      * D_MODEL + h * 8;
    const __bf16* b1 = WT + (size_t)(col0 + 16 + m) * D_MODEL + h * 8;
#pragma unroll
    for (int k0 = 0; k0 < D_MODEL; k0 += 32) {
        __builtin_prefetch(a0 + k0 + 64, 0, 1);     // global_prefetch_b8
        __builtin_prefetch(a1 + k0 + 64, 0, 1);
        Frag16 fa0, fa1, fb0, fb1;
        fa0.h[0] = *(const v8bf*)(a0 + k0);
        fa0.h[1] = *(const v8bf*)(a0 + k0 + 16);
        fa1.h[0] = *(const v8bf*)(a1 + k0);
        fa1.h[1] = *(const v8bf*)(a1 + k0 + 16);
        fb0.h[0] = *(const v8bf*)(b0 + k0);
        fb0.h[1] = *(const v8bf*)(b0 + k0 + 16);
        fb1.h[0] = *(const v8bf*)(b1 + k0);
        fb1.h[1] = *(const v8bf*)(b1 + k0 + 16);
        acc[0][0] = __builtin_amdgcn_wmma_f32_16x16x32_bf16(
                        false, fa0.v, false, fb0.v, (short)0, acc[0][0], false, false);
        acc[0][1] = __builtin_amdgcn_wmma_f32_16x16x32_bf16(
                        false, fa0.v, false, fb1.v, (short)0, acc[0][1], false, false);
        acc[1][0] = __builtin_amdgcn_wmma_f32_16x16x32_bf16(
                        false, fa1.v, false, fb0.v, (short)0, acc[1][0], false, false);
        acc[1][1] = __builtin_amdgcn_wmma_f32_16x16x32_bf16(
                        false, fa1.v, false, fb1.v, (short)0, acc[1][1], false, false);
    }
}

// in-projection: U = A @ Win + bin   (f32 out)
__global__ __launch_bounds__(256) void k_gemm_in(const __bf16* __restrict__ A,
                                                 const __bf16* __restrict__ WT,
                                                 const float* __restrict__ bias,
                                                 float* __restrict__ U) {
    const int tid = threadIdx.x, wave = tid >> 5, lane = tid & 31;
    const int row0 = blockIdx.x * 128 + (wave & 3) * 32;
    const int col0 = blockIdx.y * 64 + (wave >> 2) * 32;
    v8f acc[2][2] = {};
    gemm_core(A, WT, row0, col0, lane, acc);
    const int n0 = col0 + (lane & 15);
    const int h = lane >> 4;
#pragma unroll
    for (int ni = 0; ni < 2; ++ni) {
        const float bb = bias[n0 + 16 * ni];
#pragma unroll
        for (int mi = 0; mi < 2; ++mi) {
#pragma unroll
            for (int v = 0; v < 8; ++v) {
                const size_t o =
                    (size_t)(row0 + 16 * mi + v + 8 * h) * D_MODEL + n0 + 16 * ni;
                U[o] = acc[mi][ni][v] + bb;
            }
        }
    }
}

// out-projection + bias + exact GELU + residual  (f32 out)
__global__ __launch_bounds__(256) void k_gemm_out(const __bf16* __restrict__ A,
                                                  const __bf16* __restrict__ WT,
                                                  const float* __restrict__ bias,
                                                  const float* __restrict__ xres,
                                                  float* __restrict__ out) {
    const int tid = threadIdx.x, wave = tid >> 5, lane = tid & 31;
    const int row0 = blockIdx.x * 128 + (wave & 3) * 32;
    const int col0 = blockIdx.y * 64 + (wave >> 2) * 32;
    v8f acc[2][2] = {};
    gemm_core(A, WT, row0, col0, lane, acc);
    const int n0 = col0 + (lane & 15);
    const int h = lane >> 4;
#pragma unroll
    for (int ni = 0; ni < 2; ++ni) {
        const float bb = bias[n0 + 16 * ni];
#pragma unroll
        for (int mi = 0; mi < 2; ++mi) {
#pragma unroll
            for (int v = 0; v < 8; ++v) {
                const size_t o =
                    (size_t)(row0 + 16 * mi + v + 8 * h) * D_MODEL + n0 + 16 * ni;
                float g = acc[mi][ni][v] + bb;
                g = 0.5f * g * (1.0f + erff(g * 0.70710678118654752f));
                out[o] = xres[o] + g;
            }
        }
    }
}

// ---------------------------------------------------------------------------
// FFT helpers: 8192-pt radix-2 in LDS, 256 threads.
//   fwd: DIF, natural in -> bit-reversed out
//   inv: DIT, bit-reversed in -> natural out (caller scales by 1/8192)
// ---------------------------------------------------------------------------
__device__ __forceinline__ void fft_dif_fwd(float* __restrict__ re,
                                            float* __restrict__ im, int tid) {
    for (int sh = 12; sh >= 0; --sh) {
        const int mh = 1 << sh;
        const float base = -6.283185307179586f / (float)(mh << 1);
        for (int j = tid; j < HFFT; j += 256) {
            const int t  = j & (mh - 1);
            const int i0 = ((j >> sh) << (sh + 1)) | t;
            const int i1 = i0 + mh;
            float ur = re[i0], ui = im[i0];
            float vr = re[i1], vi = im[i1];
            re[i0] = ur + vr;
            im[i0] = ui + vi;
            float dr = ur - vr, di = ui - vi;
            float s, c;
            __sincosf(base * (float)t, &s, &c);
            re[i1] = dr * c - di * s;
            im[i1] = dr * s + di * c;
        }
        __syncthreads();
    }
}

__device__ __forceinline__ void fft_dit_inv(float* __restrict__ re,
                                            float* __restrict__ im, int tid) {
    for (int sh = 0; sh <= 12; ++sh) {
        const int mh = 1 << sh;
        const float base = 6.283185307179586f / (float)(mh << 1);
        for (int j = tid; j < HFFT; j += 256) {
            const int t  = j & (mh - 1);
            const int i0 = ((j >> sh) << (sh + 1)) | t;
            const int i1 = i0 + mh;
            float s, c;
            __sincosf(base * (float)t, &s, &c);
            float xr = re[i1], xi = im[i1];
            float vr = xr * c - xi * s;
            float vi = xr * s + xi * c;
            float ur = re[i0], ui = im[i0];
            re[i0] = ur + vr;
            im[i0] = ui + vi;
            re[i1] = ur - vr;
            im[i1] = ui - vi;
        }
        __syncthreads();
    }
}

// ---------------------------------------------------------------------------
// S4 kernel generation + forward FFT of bidirectional kernel (per chunk).
//   k[l] = sum_n e^{lr*l} * (a_n cos(li*l) - b_n sin(li*l)), a=Cre*B, b=Cim*B
//   k2 = [k ; flip(k)] -> DIF FFT -> KF[c][0..8191] (bit-reversed order)
// ---------------------------------------------------------------------------
__global__ __launch_bounds__(256) void k_s4_kernel_fft(
        const float* __restrict__ Lre, const float* __restrict__ Lim,
        const float* __restrict__ Bm,  const float* __restrict__ Cre,
        const float* __restrict__ Cim, float2* __restrict__ KF) {
    __shared__ float sre[NFFT];
    __shared__ float sim[NFFT];
    const int c = blockIdx.x, tid = threadIdx.x;
    const float* lr = Lre + (size_t)c * D_STATE;
    const float* li = Lim + (size_t)c * D_STATE;
    const float* bm = Bm  + (size_t)c * D_STATE;
    const float* cr = Cre + (size_t)c * D_STATE;
    const float* ci = Cim + (size_t)c * D_STATE;
    for (int l = tid; l < LSEQ; l += 256) {
        const float fl = (float)l;
        float acc = 0.f;
        for (int n = 0; n < D_STATE; ++n) {
            const float e = __expf(lr[n] * fl);
            float s, co;
            __sincosf(li[n] * fl, &s, &co);
            acc += e * (cr[n] * bm[n] * co - ci[n] * bm[n] * s);
        }
        sre[l] = acc;            // k[l]
        sre[NFFT - 1 - l] = acc; // k2[4096+j] = k[4095-j]
        sim[l] = 0.f;
        sim[NFFT - 1 - l] = 0.f;
    }
    __syncthreads();
    fft_dif_fwd(sre, sim, tid);
    float2* kf = KF + (size_t)c * NFFT;
    for (int j = tid; j < NFFT; j += 256) kf[j] = make_float2(sre[j], sim[j]);
}

// ---------------------------------------------------------------------------
// Per-(b,d) FFT convolution: gather u[b,:,d] (2KB stride) straight into LDS
// with GLOBAL_LOAD_ASYNC_TO_LDS_B32 (ASYNCcnt), fwd FFT, pointwise * KF[c]
// (bit-reversed order on both sides), inverse FFT, +D*u skip, emit bf16.
// ---------------------------------------------------------------------------
__global__ __launch_bounds__(256) void k_conv(const float* __restrict__ U,
                                              const float2* __restrict__ KF,
                                              const float* __restrict__ Dvec,
                                              __bf16* __restrict__ Ybf) {
    __shared__ float sre[NFFT];
    __shared__ float sim[NFFT];
    const int tid = threadIdx.x;
    const int b = blockIdx.x >> 9;
    const int d = blockIdx.x & 511;
    const int c = d >> 2;
    const float* up = U + (size_t)b * LSEQ * D_MODEL + d;
    // async gather: per-lane global addr -> per-lane LDS offset (addr[31:0])
    const unsigned lds_base = (unsigned)(uintptr_t)(&sre[0]);
    for (int t = tid; t < LSEQ; t += 256) {
        const unsigned lds_off = lds_base + (unsigned)(t * 4);
        const unsigned long long ga =
            (unsigned long long)(uintptr_t)(up + (size_t)t * D_MODEL);
        asm volatile("global_load_async_to_lds_b32 %0, %1, off"
                     :: "v"(lds_off), "v"(ga) : "memory");
        sim[t] = 0.f;
        sre[t + LSEQ] = 0.f;
        sim[t + LSEQ] = 0.f;
    }
    asm volatile("s_wait_asynccnt 0" ::: "memory");
    __syncthreads();
    fft_dif_fwd(sre, sim, tid);
    const float2* kf = KF + (size_t)c * NFFT;
    for (int j = tid; j < NFFT; j += 256) {
        const float2 w = kf[j];
        const float xr = sre[j], xi = sim[j];
        sre[j] = xr * w.x - xi * w.y;
        sim[j] = xr * w.y + xi * w.x;
    }
    __syncthreads();
    fft_dit_inv(sre, sim, tid);
    const float dv = Dvec[c];
    for (int t = tid; t < LSEQ; t += 256) {
        const float y = sre[t] * (1.0f / (float)NFFT) + up[(size_t)t * D_MODEL] * dv;
        Ybf[((size_t)b * LSEQ + t) * D_MODEL + d] = (__bf16)y;
    }
}

// ---------------------------------------------------------------------------
// Host launcher
// ---------------------------------------------------------------------------
extern "C" void kernel_launch(void* const* d_in, const int* in_sizes, int n_in,
                              void* d_out, int out_size, void* d_ws, size_t ws_size,
                              hipStream_t stream) {
    const float* x     = (const float*)d_in[0];
    const float* ln_g  = (const float*)d_in[1];
    const float* ln_b  = (const float*)d_in[2];
    const float* Win   = (const float*)d_in[3];
    const float* bin   = (const float*)d_in[4];
    const float* Wout  = (const float*)d_in[5];
    const float* bout  = (const float*)d_in[6];
    const float* Lre   = (const float*)d_in[7];
    const float* Lim   = (const float*)d_in[8];
    const float* Bm    = (const float*)d_in[9];
    const float* Cre   = (const float*)d_in[10];
    const float* Cim   = (const float*)d_in[11];
    const float* Dvec  = (const float*)d_in[12];
    float* out = (float*)d_out;

    // workspace carve-out (256B aligned)
    char* p = (char*)d_ws;
    size_t off = 0;
    auto take = [&](size_t bytes) -> char* {
        char* r = p + off;
        off = (off + bytes + 255) & ~(size_t)255;
        return r;
    };
    __bf16* Zbf   = (__bf16*)take((size_t)NTOK * D_MODEL * 2);
    __bf16* WinT  = (__bf16*)take((size_t)D_MODEL * D_MODEL * 2);
    __bf16* WoutT = (__bf16*)take((size_t)D_MODEL * D_MODEL * 2);
    float*  U     = (float*) take((size_t)NTOK * D_MODEL * 4);
    __bf16* Ybf   = (__bf16*)take((size_t)NTOK * D_MODEL * 2);
    float2* KF    = (float2*)take((size_t)NCHUNK * NFFT * 8);
    (void)ws_size; (void)n_in; (void)in_sizes; (void)out_size;

    // 1. transpose + bf16-convert weights
    k_weight_t<<<(D_MODEL * D_MODEL) / 256, 256, 0, stream>>>(Win, WinT);
    k_weight_t<<<(D_MODEL * D_MODEL) / 256, 256, 0, stream>>>(Wout, WoutT);
    // 2. LayerNorm -> bf16
    k_layernorm<<<NTOK, 256, 0, stream>>>(x, ln_g, ln_b, Zbf);
    // 3. in-projection (WMMA)
    k_gemm_in<<<dim3(NTOK / 128, D_MODEL / 64), 256, 0, stream>>>(Zbf, WinT, bin, U);
    // 4. S4 kernel spectra (128 chunks)
    k_s4_kernel_fft<<<NCHUNK, 256, 0, stream>>>(Lre, Lim, Bm, Cre, Cim, KF);
    // 5. FFT convolution per (batch, channel)
    k_conv<<<BSZ * D_MODEL, 256, 0, stream>>>(U, KF, Dvec, Ybf);
    // 6. out-projection + GELU + residual (WMMA)
    k_gemm_out<<<dim3(NTOK / 128, D_MODEL / 64), 256, 0, stream>>>(Ybf, WoutT, bout, x, out);
}
